// SimpleAttention_37898791420287
// MI455X (gfx1250) — matrix operand order repaired and checked by reading, not compiled, not run on hardware
//
#include <hip/hip_runtime.h>
#include <stdint.h>

#define B_  4
#define L_  8192
#define D_  512
#define H_  8
#define DK_ 64
#define E_  65
#define BL_ (B_*L_)
#define FCW_COLS (D_ + H_)   // 520

typedef __attribute__((ext_vector_type(16))) __bf16          v16bf;
typedef __attribute__((ext_vector_type(8)))  float           v8f;
typedef __attribute__((ext_vector_type(16))) unsigned short  v16u;
typedef __attribute__((ext_vector_type(8)))  unsigned short  v8u;
typedef __attribute__((ext_vector_type(4)))  unsigned int    v4u;
typedef __attribute__((ext_vector_type(8)))  int             v8i;
typedef __attribute__((ext_vector_type(4)))  int             v4i;

union BF16x16 { v16u u; v16bf b; };

__device__ __forceinline__ unsigned short f2bfu(float f) {
    __bf16 h = (__bf16)f;                       // RNE truncation via hw cvt
    return __builtin_bit_cast(unsigned short, h);
}

__device__ __forceinline__ v16bf as_bf(v16u u) {
    union { v16u u; v16bf b; } x; x.u = u; return x.b;
}

// ---------------------------------------------------------------------------
// TDM: 2D tile load Global -> LDS (bf16/2-byte elements).
// D# built per CDNA5 ISA ch.8: group0 {count=1, lds_addr, global_addr, type=2},
// group1 {data_size=2B, tensor_dim0=stride, tensor_dim1=1M, tile_dim0/1,
//         tensor_dim0_stride}.  2D tensor -> groups 2/3 zero.
// This toolchain exposes the 6-arg builtin (extra zero-filled int32x8 group).
// ---------------------------------------------------------------------------
__device__ __forceinline__ void tdm_load_2d(unsigned lds_off, const void* gptr,
                                            unsigned tile_w_elems, unsigned tile_h,
                                            unsigned stride_elems) {
    unsigned long long ga = (unsigned long long)gptr;
    v4u g0;
    g0[0] = 1u;                                              // count=1, user mode
    g0[1] = lds_off;                                         // lds_addr (bytes)
    g0[2] = (unsigned)(ga & 0xFFFFFFFFu);                    // global_addr[31:0]
    g0[3] = (unsigned)((ga >> 32) & 0x01FFFFFFu) | (2u << 30); // addr[56:32], type=2
    v8i g1;
    g1[0] = (int)(1u << 16);                                 // data_size=1 -> 2 bytes
    g1[1] = (int)((stride_elems & 0xFFFFu) << 16);           // tensor_dim0[15:0]
    g1[2] = 0;                                               // dim0[31:16]=0, dim1[15:0]=0
    g1[3] = (int)(16u | (tile_w_elems << 16));               // dim1[31:16]=16 (1M rows), tile_dim0
    g1[4] = (int)(tile_h & 0xFFFFu);                         // tile_dim1, tile_dim2=0
    g1[5] = (int)stride_elems;                               // tensor_dim0_stride[31:0]
    g1[6] = 0;
    g1[7] = 0;
    v4i z4 = v4i{};
    v8i z8 = v8i{};
    __builtin_amdgcn_tensor_load_to_lds(g0, g1, z4, z4, z8, 0);
}

// ---------------------------------------------------------------------------
// Kernel 0: fp32 -> bf16 bulk convert (weights)
// ---------------------------------------------------------------------------
__global__ __launch_bounds__(256) void k_cvt(const float* __restrict__ src,
                                             unsigned short* __restrict__ dst, int n) {
    int i = blockIdx.x * 256 + threadIdx.x;
    if (i < n) dst[i] = f2bfu(src[i]);
}

// ---------------------------------------------------------------------------
// Kernel 1: projection GEMM  C[BL,512] = X @ W^T + b, optional per-head LN,
// store bf16.  One wave -> 16 rows x 64 cols (one full head strip).
// B operand (Wb) is pre-converted bf16: direct 32-byte fragment loads.
// ---------------------------------------------------------------------------
__global__ __launch_bounds__(256) void k_proj(const float* __restrict__ X,
                                              const unsigned short* __restrict__ Wb,
                                              const float* __restrict__ bias,
                                              const float* __restrict__ gam,
                                              const float* __restrict__ bet,
                                              unsigned short* __restrict__ out,
                                              int doLN) {
    const int wave = threadIdx.x >> 5;
    const int lane = threadIdx.x & 31;
    const int hi   = lane >> 4;
    const int lo   = lane & 15;
    const int m0   = (blockIdx.x * 8 + wave) * 16;
    const int head = blockIdx.y;
    const int n0   = head * DK_;

    v8f acc[4] = {v8f{}, v8f{}, v8f{}, v8f{}};

    const float* arow = X + (size_t)(m0 + lo) * D_;

    for (int kb = 0; kb < D_; kb += 32) {
        __builtin_prefetch(arow + kb + 64, 0, 0);
        // A fragment (16x32 bf16): K = kb + hi*16 + j
        BF16x16 a;
        const float4* p0 = (const float4*)(arow + kb + hi * 16);
        const float4* p1 = (const float4*)(arow + kb + hi * 16 + 8);
        float4 a0 = p0[0], a1 = p0[1], a2 = p1[0], a3 = p1[1];
        a.b[0]  = (__bf16)a0.x; a.b[1]  = (__bf16)a0.y; a.b[2]  = (__bf16)a0.z; a.b[3]  = (__bf16)a0.w;
        a.b[4]  = (__bf16)a1.x; a.b[5]  = (__bf16)a1.y; a.b[6]  = (__bf16)a1.z; a.b[7]  = (__bf16)a1.w;
        a.b[8]  = (__bf16)a2.x; a.b[9]  = (__bf16)a2.y; a.b[10] = (__bf16)a2.z; a.b[11] = (__bf16)a2.w;
        a.b[12] = (__bf16)a3.x; a.b[13] = (__bf16)a3.y; a.b[14] = (__bf16)a3.z; a.b[15] = (__bf16)a3.w;

#pragma unroll
        for (int t = 0; t < 4; ++t) {
            // B fragment (32x16 bf16): row n = n0+t*16+lo of W, K = kb + hi*16 + j
            v16u bu = *(const v16u*)(Wb + (size_t)(n0 + t * 16 + lo) * D_ + kb + hi * 16);
            acc[t] = __builtin_amdgcn_wmma_f32_16x16x32_bf16(
                false, a.b, false, as_bf(bu), (short)0, acc[t], false, false);
        }
    }

    float bb[4];
#pragma unroll
    for (int t = 0; t < 4; ++t) bb[t] = bias[n0 + t * 16 + lo];
#pragma unroll
    for (int t = 0; t < 4; ++t)
#pragma unroll
        for (int r = 0; r < 8; ++r) acc[t][r] += bb[t];

    if (doLN) {
        float g4[4], b4[4];
#pragma unroll
        for (int t = 0; t < 4; ++t) {
            g4[t] = gam[n0 + t * 16 + lo];
            b4[t] = bet[n0 + t * 16 + lo];
        }
#pragma unroll
        for (int r = 0; r < 8; ++r) {
            float s  = acc[0][r] + acc[1][r] + acc[2][r] + acc[3][r];
            float s2 = acc[0][r] * acc[0][r] + acc[1][r] * acc[1][r]
                     + acc[2][r] * acc[2][r] + acc[3][r] * acc[3][r];
#pragma unroll
            for (int m = 1; m < 16; m <<= 1) {
                s  += __shfl_xor(s,  m, 16);
                s2 += __shfl_xor(s2, m, 16);
            }
            float mean = s * (1.0f / 64.0f);
            float var  = s2 * (1.0f / 64.0f) - mean * mean;   // biased (torch LN)
            float inv  = rsqrtf(var + 1e-5f);
#pragma unroll
            for (int t = 0; t < 4; ++t)
                acc[t][r] = (acc[t][r] - mean) * inv * g4[t] + b4[t];
        }
    }

#pragma unroll
    for (int t = 0; t < 4; ++t)
#pragma unroll
        for (int r = 0; r < 8; ++r) {
            int m = m0 + r + hi * 8;
            out[(size_t)m * D_ + n0 + t * 16 + lo] = f2bfu(acc[t][r]);
        }
}

// ---------------------------------------------------------------------------
// Kernel 2: attn[b,h] = K'^T V' / L, K'=[pos|Kn], V'=[pos|Vn]  ([L,65] each).
// One workgroup per (b,h).  L is processed in 128-row chunks staged in LDS
// by the Tensor Data Mover (wave 0 issues 2 descriptors, waits TENSORcnt,
// workgroup barrier).  Each wave owns up to 4 of the 25 16x16 output tiles.
// ---------------------------------------------------------------------------
__global__ __launch_bounds__(256) void k_attn(const unsigned short* __restrict__ Kb,
                                              const unsigned short* __restrict__ Vb,
                                              const float* __restrict__ pos,
                                              float* __restrict__ attn_out) {
    __shared__ unsigned short smem[128 * 64 * 2 + 128];
    unsigned short* ldsK   = smem;                 // LDS byte offset 0
    unsigned short* ldsV   = smem + 128 * 64;      // LDS byte offset 16384
    unsigned short* posLds = smem + 2 * 128 * 64;  // bf16 pos column

    const int wave = threadIdx.x >> 5;
    const int lane = threadIdx.x & 31;
    const int hi   = lane >> 4;
    const int lo   = lane & 15;
    const int b    = blockIdx.x >> 3;
    const int h    = blockIdx.x & 7;

    const size_t gbase = (size_t)b * L_ * D_ + (size_t)h * DK_;
    const float* pb    = pos + (size_t)b * L_;

    v8f acc[4] = {v8f{}, v8f{}, v8f{}, v8f{}};

    for (int l0 = 0; l0 < L_; l0 += 128) {
        __syncthreads();                           // LDS reuse fence
        if (wave == 0) {
            tdm_load_2d(0u,     Kb + gbase + (size_t)l0 * D_, DK_, 128, D_);
            tdm_load_2d(16384u, Vb + gbase + (size_t)l0 * D_, DK_, 128, D_);
            __builtin_amdgcn_s_wait_tensorcnt((short)0);
        }
        if (threadIdx.x >= 128) {                  // waves 4..7 stage pos column
            int t = threadIdx.x - 128;
            posLds[t] = f2bfu(pb[l0 + t]);
        }
        __syncthreads();

#pragma unroll
        for (int i = 0; i < 4; ++i) {
            int t = wave + 8 * i;
            if (t >= 25) break;
            const int d0 = (t / 5) * 16, e0 = (t % 5) * 16;
            const int d = d0 + lo;                 // A column of K' (M index)
            const int e = e0 + lo;                 // B column of V' (N index)
#pragma unroll
            for (int lb = 0; lb < 128; lb += 32) {
                BF16x16 a, bf;
#pragma unroll
                for (int j = 0; j < 16; ++j) {
                    int la = lb + hi * 16 + j;                            // A K idx
                    unsigned short av;
                    if (d == 0)        av = posLds[la];
                    else if (d <= DK_) av = ldsK[la * DK_ + (d - 1)];
                    else               av = 0;
                    a.u[j] = av;

                    int lv = lb + hi * 16 + j;                            // B K idx
                    unsigned short bv;
                    if (e == 0)        bv = posLds[lv];
                    else if (e <= DK_) bv = ldsV[lv * DK_ + (e - 1)];
                    else               bv = 0;
                    bf.u[j] = bv;
                }
                acc[i] = __builtin_amdgcn_wmma_f32_16x16x32_bf16(
                    false, a.b, false, bf.b, (short)0, acc[i], false, false);
            }
        }
    }

    const float scl = 1.0f / (float)L_;
#pragma unroll
    for (int i = 0; i < 4; ++i) {
        int t = wave + 8 * i;
        if (t >= 25) break;
        const int d0 = (t / 5) * 16, e0 = (t % 5) * 16;
#pragma unroll
        for (int r = 0; r < 8; ++r) {
            int dr = d0 + r + hi * 8;
            int ec = e0 + lo;
            if (dr < E_ && ec < E_)
                attn_out[(((size_t)(b * H_ + h)) * E_ + dr) * E_ + ec] = acc[i][r] * scl;
        }
    }
}

// ---------------------------------------------------------------------------
// Kernel 3a: MbigT[b][n][kk] = (attn_h @ fcW_h^T)[e,n], kk = h*64 + (e-1)
// ---------------------------------------------------------------------------
__global__ __launch_bounds__(256) void k_mproj(const float* __restrict__ attn,
                                               const float* __restrict__ fcW,
                                               unsigned short* __restrict__ MbigT) {
    int idx = blockIdx.x * 256 + threadIdx.x;     // b*2^18 + n*2^9 + kk
    int b  = idx >> 18;
    int n  = (idx >> 9) & 511;
    int kk = idx & 511;
    int h  = kk >> 6;
    int e  = (kk & 63) + 1;
    const float* arow = attn + (((size_t)(b * H_ + h)) * E_ + e) * E_;
    const float* wrow = fcW + (size_t)n * FCW_COLS + h * E_;
    float s = 0.0f;
#pragma unroll 5
    for (int j = 0; j < E_; ++j) s += arow[j] * wrow[j];
    MbigT[((size_t)b * D_ + n) * D_ + kk] = f2bfu(s);
}

// Kernel 3b: S[b][n] = sum_h (attn_h[0,:] @ fcW_h^T)[n]  (pos rank-1 term)
__global__ __launch_bounds__(256) void k_sproj(const float* __restrict__ attn,
                                               const float* __restrict__ fcW,
                                               float* __restrict__ S) {
    int idx = blockIdx.x * 256 + threadIdx.x;     // b*512 + n
    int b = idx >> 9;
    int n = idx & 511;
    float s = 0.0f;
    for (int h = 0; h < H_; ++h) {
        const float* arow = attn + (((size_t)(b * H_ + h)) * E_) * E_;
        const float* wrow = fcW + (size_t)n * FCW_COLS + h * E_;
        for (int j = 0; j < E_; ++j) s += arow[j] * wrow[j];
    }
    S[b * D_ + n] = s;
}

// ---------------------------------------------------------------------------
// Kernel 4: att_out[b] = Q[b] @ Mbig[b] + pos[b] (x) S[b] + fcb
// ---------------------------------------------------------------------------
__global__ __launch_bounds__(256) void k_final(const unsigned short* __restrict__ Qb,
                                               const unsigned short* __restrict__ MbigT,
                                               const float* __restrict__ S,
                                               const float* __restrict__ pos,
                                               const float* __restrict__ fcb,
                                               float* __restrict__ out) {
    const int wave = threadIdx.x >> 5;
    const int lane = threadIdx.x & 31;
    const int hi   = lane >> 4;
    const int lo   = lane & 15;

    const int gw  = blockIdx.x * 8 + wave;        // 0..65535
    const int b   = gw >> 14;
    const int rem = gw & 16383;
    const int m0  = (rem >> 5) * 16;              // 512 m-tiles
    const int n0  = (rem & 31) * 16;              // 32 n-tiles

    const unsigned short* qrow = Qb + (size_t)(b * L_ + m0 + lo) * D_;
    const unsigned short* mrow = MbigT + ((size_t)b * D_ + n0 + lo) * D_;

    v8f acc = v8f{};

    for (int kb = 0; kb < D_; kb += 32) {
        v8u qlo = *(const v8u*)(qrow + kb + hi * 16);
        v8u qhi = *(const v8u*)(qrow + kb + hi * 16 + 8);
        v16u au = __builtin_shufflevector(qlo, qhi,
                    0, 1, 2, 3, 4, 5, 6, 7, 8, 9, 10, 11, 12, 13, 14, 15);
        v16u bu = *(const v16u*)(mrow + kb + hi * 16);
        acc = __builtin_amdgcn_wmma_f32_16x16x32_bf16(
            false, as_bf(au), false, as_bf(bu), (short)0, acc, false, false);
    }

    const float sv = S[b * D_ + n0 + lo];
    const float bb = fcb[n0 + lo];
    const float* pb = pos + (size_t)b * L_;
#pragma unroll
    for (int r = 0; r < 8; ++r) {
        int m = m0 + r + hi * 8;
        out[((size_t)b * L_ + m) * D_ + n0 + lo] = acc[r] + pb[m] * sv + bb;
    }
}

// ---------------------------------------------------------------------------
extern "C" void kernel_launch(void* const* d_in, const int* in_sizes, int n_in,
                              void* d_out, int out_size, void* d_ws, size_t ws_size,
                              hipStream_t stream) {
    const float* query = (const float*)d_in[0];
    const float* key   = (const float*)d_in[1];
    const float* value = (const float*)d_in[2];
    const float* pos   = (const float*)d_in[3];
    const float* Wq    = (const float*)d_in[4];
    const float* bq    = (const float*)d_in[5];
    const float* Wk    = (const float*)d_in[6];
    const float* bk    = (const float*)d_in[7];
    const float* Wv    = (const float*)d_in[8];
    const float* bv    = (const float*)d_in[9];
    const float* gK    = (const float*)d_in[10];
    const float* betaK = (const float*)d_in[11];
    const float* gV    = (const float*)d_in[12];
    const float* betaV = (const float*)d_in[13];
    const float* fcW   = (const float*)d_in[14];
    const float* fcb   = (const float*)d_in[15];

    float* att_out  = (float*)d_out;                                // [B,L,D]
    float* attn_out = att_out + (size_t)B_ * L_ * D_;               // [B,H,65,65]

    unsigned short* Qb    = (unsigned short*)d_ws;                  // bf16 [BL,512]
    unsigned short* Kb    = Qb + (size_t)BL_ * D_;
    unsigned short* Vb    = Kb + (size_t)BL_ * D_;
    unsigned short* MbigT = Vb + (size_t)BL_ * D_;                  // bf16 [B,512,512]
    float*          S     = (float*)(MbigT + (size_t)B_ * D_ * D_); // f32 [B,512]
    unsigned short* Wbq   = (unsigned short*)(S + (size_t)B_ * D_); // bf16 [512,512] x3
    unsigned short* Wbk   = Wbq + (size_t)D_ * D_;
    unsigned short* Wbv   = Wbk + (size_t)D_ * D_;

    dim3 blk(256);
    const int nW = D_ * D_;   // 262144

    // Stage 0: one-time bf16 conversion of the weight matrices
    k_cvt<<<nW / 256, blk, 0, stream>>>(Wq, Wbq, nW);
    k_cvt<<<nW / 256, blk, 0, stream>>>(Wk, Wbk, nW);
    k_cvt<<<nW / 256, blk, 0, stream>>>(Wv, Wbv, nW);

    // Stage 1: projections (+ per-head LN for K, V), store bf16
    k_proj<<<dim3(256, H_), blk, 0, stream>>>(query, Wbq, bq, nullptr, nullptr, Qb, 0);
    k_proj<<<dim3(256, H_), blk, 0, stream>>>(key,   Wbk, bk, gK, betaK,       Kb, 1);
    k_proj<<<dim3(256, H_), blk, 0, stream>>>(value, Wbv, bv, gV, betaV,       Vb, 1);

    // Stage 2: attn = K'^T V' / L  (TDM-staged LDS tiles)
    k_attn<<<B_ * H_, blk, 0, stream>>>(Kb, Vb, pos, attn_out);

    // Stage 3: fold attn into fcW  ->  MbigT (bf16) and pos rank-1 vector S
    k_mproj<<<4096, blk, 0, stream>>>(attn_out, fcW, MbigT);
    k_sproj<<<8,    blk, 0, stream>>>(attn_out, fcW, S);

    // Stage 4: att_out = Q @ Mbig + pos (x) S + fcb
    k_final<<<8192, blk, 0, stream>>>(Qb, MbigT, S, pos, fcb, att_out);
}
